// MultiQueryAttention_33852932227544
// MI455X (gfx1250) — matrix operand order repaired and checked by reading, not compiled, not run on hardware
//
#include <hip/hip_runtime.h>

#define S_LEN    2048
#define D_MODELC 2048
#define NUM_H    16
#define HEAD_D   128
#define BATCH    2

typedef __attribute__((ext_vector_type(16))) __bf16       v16bf;
typedef __attribute__((ext_vector_type(8)))  float        v8f;
typedef __attribute__((ext_vector_type(8)))  unsigned int u32x8;
typedef __attribute__((ext_vector_type(4)))  unsigned int uint32x4;
typedef __attribute__((ext_vector_type(8)))  int          int32x8;
typedef __attribute__((ext_vector_type(4)))  int          int32x4;

#if __has_builtin(__builtin_amdgcn_tensor_load_to_lds) && \
    __has_builtin(__builtin_amdgcn_s_wait_tensorcnt)
#define ATTN_USE_TDM 1
#endif

struct BF16Frag { uint4 lo; uint4 hi; };   // 16 bf16 = one WMMA A/B operand per lane

__device__ __forceinline__ unsigned short f2bf(float f) {
  unsigned int u = __builtin_bit_cast(unsigned int, f);
  u += 0x7fffu + ((u >> 16) & 1u);          // round-to-nearest-even
  return (unsigned short)(u >> 16);
}

__device__ __forceinline__ v8f zero8() {
  v8f z = {0.f, 0.f, 0.f, 0.f, 0.f, 0.f, 0.f, 0.f};
  return z;
}

__device__ __forceinline__ v8f wmma_bf16(const BF16Frag& a, const BF16Frag& b, v8f c) {
  v16bf av = __builtin_bit_cast(v16bf, a);
  v16bf bv = __builtin_bit_cast(v16bf, b);
  return __builtin_amdgcn_wmma_f32_16x16x32_bf16(false, av, false, bv,
                                                 (short)0, c, false, false);
}

#ifdef ATTN_USE_TDM
// Issue a 2D TDM tile load (bf16 elements) per CDNA5 ISA D# layout (§8).
// stride0 = elements between consecutive dim1 lines; optional LDS row padding.
// 6-arg builtin: (g0, g1, g2, g3, g4, cpol); trailing int32x8 zero for 2D.
__device__ __forceinline__ void tdm_load_2d(unsigned lds_addr, const void* gaddr,
                                            unsigned tensor_d0, unsigned tensor_d1,
                                            unsigned tile_d0, unsigned tile_d1,
                                            unsigned long long stride0,
                                            unsigned pad_interval_code,
                                            unsigned pad_amount_code) {
  unsigned long long ga = (unsigned long long)(size_t)gaddr;
  uint32x4 g0;
  g0[0] = 1u;                                     // count=1 (valid), user mode
  g0[1] = lds_addr;                               // lds_addr [63:32]
  g0[2] = (unsigned)ga;                           // global_addr [95:64]
  g0[3] = (unsigned)(ga >> 32) | (2u << 30);      // global_addr hi | type=2
  int32x8 g1;
  unsigned dw0 = (1u << 16)                       // data_size=1 -> 2 bytes
               | (1u << 20)                       // pad_enable
               | (pad_interval_code << 22)
               | (pad_amount_code << 25);
  g1[0] = (int)dw0;                               // workgroup_mask=0 (no cluster)
  g1[1] = (int)((tensor_d0 & 0xffffu) << 16);     // tensor_dim0[15:0] @ bits 63:48
  g1[2] = (int)((tensor_d0 >> 16) | ((tensor_d1 & 0xffffu) << 16));
  g1[3] = (int)((tensor_d1 >> 16) | (tile_d0 << 16));   // tile_dim0 @ 127:112
  g1[4] = (int)tile_d1;                           // tile_dim1; tile_dim2=0
  g1[5] = (int)(unsigned)stride0;                 // tensor_dim0_stride[31:0]
  g1[6] = (int)((unsigned)(stride0 >> 32) & 0xffffu);   // stride0 hi; dim1_stride=0
  g1[7] = 0;
  int32x4 z4 = {0, 0, 0, 0};                      // groups 2/3: 2D tile, unused
  int32x8 z8 = {0, 0, 0, 0, 0, 0, 0, 0};
  __builtin_amdgcn_tensor_load_to_lds(g0, g1, z4, z4, z8, 0);
}
#endif

// ---------------------------------------------------------------- convert ---
__global__ void f32_to_bf16_kernel(const float* __restrict__ in,
                                   unsigned short* __restrict__ out, int n) {
  int i = blockIdx.x * blockDim.x + threadIdx.x;
  if (i < n) out[i] = f2bf(in[i]);
}

// ------------------------------------------------------------------- GEMM ---
// C[M,N] = alpha * A[M,K] @ W[N,K]^T.  out_mode: 0=f32, 1=bf16, 2=bf16 C^T.
// One wave computes a 32x64 tile (8 accumulators); fragments double-buffered.
// launch_bounds(128,1): give RA the full VGPR file (no spills) — for a
// compute-bound WMMA loop, pipelined fragments beat occupancy.
__global__ void __launch_bounds__(128, 1)
gemm_bf16_wmma(const unsigned short* __restrict__ A,
               const unsigned short* __restrict__ W,
               void* __restrict__ Cout,
               int M, int N, int K, float alpha, int out_mode) {
  int wave = blockIdx.x * (blockDim.x >> 5) + (threadIdx.x >> 5);
  int lane = threadIdx.x & 31;
  int half = lane >> 4, l15 = lane & 15;
  int ntiles = N >> 6;
  int mt = wave / ntiles;
  int nt = wave % ntiles;
  if (mt >= (M >> 5)) return;

  const unsigned short* a0 = A + (size_t)(mt * 32 + l15) * K;
  const unsigned short* a1 = a0 + (size_t)16 * K;
  const unsigned short* wp0 = W + (size_t)(nt * 64 + l15) * K;

  v8f acc[8];
#pragma unroll
  for (int i = 0; i < 8; ++i) acc[i] = zero8();

  BF16Frag aX[2], bX[4], aY[2], bY[4];

  auto load_set = [&](BF16Frag* af, BF16Frag* bf, int k0) {
    const unsigned short* p = a0 + k0 + 8 * half;           // A: 2x 16B @ +0/+32B
    af[0].lo = *(const uint4*)p;       af[0].hi = *(const uint4*)(p + 16);
    p = a1 + k0 + 8 * half;
    af[1].lo = *(const uint4*)p;       af[1].hi = *(const uint4*)(p + 16);
#pragma unroll
    for (int j = 0; j < 4; ++j) {                           // B: 2x 16B @ +0/+32B
      const unsigned short* q = wp0 + (size_t)(j * 16) * K + k0 + 8 * half;
      bf[j].lo = *(const uint4*)q;     bf[j].hi = *(const uint4*)(q + 16);
    }
  };
  auto mma_set = [&](const BF16Frag* af, const BF16Frag* bf) {
#pragma unroll
    for (int j = 0; j < 4; ++j) {
      acc[j]     = wmma_bf16(af[0], bf[j], acc[j]);
      acc[4 + j] = wmma_bf16(af[1], bf[j], acc[4 + j]);
    }
  };

  // branch-free steady state (K is a multiple of 64)
  load_set(aX, bX, 0);
  int k0 = 0;
  for (; k0 + 64 < K; k0 += 64) {
    load_set(aY, bY, k0 + 32);
    mma_set(aX, bX);
    load_set(aX, bX, k0 + 64);
    mma_set(aY, bY);
  }
  load_set(aY, bY, k0 + 32);   // epilogue: last two k-chunks
  mma_set(aX, bX);
  mma_set(aY, bY);

#pragma unroll
  for (int mi = 0; mi < 2; ++mi)
#pragma unroll
    for (int j = 0; j < 4; ++j)
#pragma unroll
      for (int r = 0; r < 8; ++r) {
        float v  = acc[mi * 4 + j][r] * alpha;
        int rowm = mt * 32 + mi * 16 + r + 8 * half;
        int col  = nt * 64 + j * 16 + l15;
        if (out_mode == 0)
          ((float*)Cout)[(size_t)rowm * N + col] = v;
        else if (out_mode == 1)
          ((unsigned short*)Cout)[(size_t)rowm * N + col] = f2bf(v);
        else
          ((unsigned short*)Cout)[(size_t)col * M + rowm] = f2bf(v);
      }
}

// -------------------------------------------------------- flash attention ---
// One wave = one 16-row Q tile of one (batch, head). The 4 waves of a block
// are 4 adjacent Q tiles of the SAME (b,h) and (MQA) share one K/V stream:
// K/V chunks are staged into LDS by the Tensor Data Mover (double-buffered,
// TENSORcnt-synced), with TDM row padding to avoid LDS bank aliasing.
// Scores are computed TRANSPOSED (T = K·Q^T) so T's C-layout is exactly the
// A-fragment layout needed for P in the P·V WMMA (no cross-lane movement).
__global__ void __launch_bounds__(128, 1)
mqa_flash_attn(const unsigned short* __restrict__ Q,
               const unsigned short* __restrict__ Kc,
               const unsigned short* __restrict__ Vt,
               unsigned short* __restrict__ O) {
  const int KROW = 136;   // K LDS row stride (128 bf16 + 16B TDM pad) in elems
  const int VROW = 40;    // V LDS row stride (32 bf16 + 16B TDM pad) in elems
  __shared__ unsigned short kbuf[2][32 * KROW];    // 2 x 8704 B
  __shared__ unsigned short vbuf[2][HEAD_D * VROW];// 2 x 10240 B

  int wave = blockIdx.x * (blockDim.x >> 5) + (threadIdx.x >> 5);
  int lane = threadIdx.x & 31;
  int mt = wave & 127;
  int h  = (wave >> 7) & 15;
  int b  = wave >> 11;
  int half = lane >> 4, l15 = lane & 15;
  const int BS = BATCH * S_LEN;

  const unsigned short* kbase = Kc + (size_t)b * S_LEN * HEAD_D;
  const unsigned short* vtbase = Vt;   // [HEAD_D][B*S], column offset b*S+kv

  // Resident Q B-fragments (d split into 4 k-chunks of 32)
  BF16Frag qf[4];
  {
    const unsigned short* qp =
        Q + (size_t)(b * S_LEN + mt * 16 + l15) * D_MODELC + h * HEAD_D;
#pragma unroll
    for (int kc = 0; kc < 4; ++kc) {
      const unsigned short* p = qp + kc * 32 + 8 * half;
      qf[kc].lo = *(const uint4*)p;
      qf[kc].hi = *(const uint4*)(p + 16);
    }
  }

  auto stage_tiles = [&](int buf, int kv0) {
#ifdef ATTN_USE_TDM
    if (threadIdx.x < 32) {   // one wave issues the DMA for the whole block
      // K chunk: 32 kv-rows x 128 d (row-contiguous); pad 16B per 256B row
      tdm_load_2d((unsigned)(size_t)&kbuf[buf][0],
                  kbase + (size_t)kv0 * HEAD_D,
                  HEAD_D, (unsigned)(BATCH * S_LEN), HEAD_D, 32,
                  (unsigned long long)HEAD_D, /*interval 2^5*8B=256B*/ 5,
                  /*amount 4 dwords=16B*/ 3);
      // V^T chunk: 128 d-rows x 32 kv (stride BS); pad 16B per 64B row
      tdm_load_2d((unsigned)(size_t)&vbuf[buf][0],
                  vtbase + (size_t)b * S_LEN + kv0,
                  (unsigned)BS, HEAD_D, 32, HEAD_D,
                  (unsigned long long)BS, /*interval 2^3*8B=64B*/ 3,
                  /*amount 16B*/ 3);
    }
#else
    // Cooperative fallback copy (same padded LDS layout)
    for (int idx = threadIdx.x; idx < 32 * (HEAD_D / 8); idx += blockDim.x) {
      int r = idx >> 4, c = idx & 15;
      ((uint4*)&kbuf[buf][r * KROW])[c] =
          ((const uint4*)(kbase + (size_t)(kv0 + r) * HEAD_D))[c];
    }
    for (int idx = threadIdx.x; idx < HEAD_D * 4; idx += blockDim.x) {
      int r = idx >> 2, c = idx & 3;
      ((uint4*)&vbuf[buf][r * VROW])[c] =
          ((const uint4*)(vtbase + (size_t)r * BS + b * S_LEN + kv0))[c];
    }
#endif
  };

  float mrow = -1e30f, lrow = 0.f;
  v8f oacc[8];
#pragma unroll
  for (int i = 0; i < 8; ++i) oacc[i] = zero8();

  stage_tiles(0, 0);
  int cur = 0;
  for (int kv0 = 0; kv0 < S_LEN; kv0 += 32, cur ^= 1) {
#ifdef ATTN_USE_TDM
    __builtin_amdgcn_s_wait_tensorcnt(0);   // issuer waits; others: no-op
#endif
    __syncthreads();                        // LDS tiles visible to all waves
    if (kv0 + 32 < S_LEN) stage_tiles(cur ^ 1, kv0 + 32);  // overlap next DMA

    // ---- T = K · Q^T : two 16x16 tiles (kv halves), 4 chained WMMAs each
    v8f t0 = zero8(), t1 = zero8();
    const unsigned short* kb = &kbuf[cur][0];
#pragma unroll
    for (int kc = 0; kc < 4; ++kc) {
      BF16Frag ka;
      const unsigned short* kp = kb + l15 * KROW + kc * 32 + 8 * half;
      ka.lo = *(const uint4*)kp;
      ka.hi = *(const uint4*)(kp + 16);
      t0 = wmma_bf16(ka, qf[kc], t0);
      kp += 16 * KROW;
      ka.lo = *(const uint4*)kp;
      ka.hi = *(const uint4*)(kp + 16);
      t1 = wmma_bf16(ka, qf[kc], t1);
    }

    // ---- online softmax over kv (per-lane q-column m = l15)
    float tmax = -1e30f;
#pragma unroll
    for (int r = 0; r < 8; ++r) tmax = fmaxf(tmax, fmaxf(t0[r], t1[r]));
    tmax = fmaxf(tmax, __shfl_xor(tmax, 16, 32));
    float mnew   = fmaxf(mrow, tmax);
    float ascale = __expf(mrow - mnew);
    float p0[8], p1[8], rsum = 0.f;
#pragma unroll
    for (int r = 0; r < 8; ++r) {
      p0[r] = __expf(t0[r] - mnew);
      p1[r] = __expf(t1[r] - mnew);
      rsum += p0[r] + p1[r];
    }
    rsum += __shfl_xor(rsum, 16, 32);
    lrow = lrow * ascale + rsum;
    mrow = mnew;

    // ---- pack P as bf16 A-fragment (layout already matches, in-lane pack)
    u32x8 pw;
#pragma unroll
    for (int i = 0; i < 4; ++i)
      pw[i] = (unsigned int)f2bf(p0[2 * i]) |
              ((unsigned int)f2bf(p0[2 * i + 1]) << 16);
#pragma unroll
    for (int i = 0; i < 4; ++i)
      pw[4 + i] = (unsigned int)f2bf(p1[2 * i]) |
                  ((unsigned int)f2bf(p1[2 * i + 1]) << 16);
    BF16Frag pf = __builtin_bit_cast(BF16Frag, pw);

    float arow[8];
#pragma unroll
    for (int r = 0; r < 8; ++r) arow[r] = __shfl(ascale, r + 8 * half, 32);

    // ---- O = O*alpha + P · V   (V fragments from LDS, ds_load_b128)
#pragma unroll
    for (int blk = 0; blk < 8; ++blk) {
      const unsigned short* vp = &vbuf[cur][(blk * 16 + l15) * VROW + 8 * half];
      BF16Frag vf;
      vf.lo = *(const uint4*)vp;
      vf.hi = *(const uint4*)(vp + 16);
      v8f c = oacc[blk];
#pragma unroll
      for (int r = 0; r < 8; ++r) c[r] *= arow[r];
      oacc[blk] = wmma_bf16(pf, vf, c);
    }
    __syncthreads();   // all reads of buf[cur] done before it is re-filled
  }

  // ---- normalize and store head output (bf16 for the O-projection GEMM)
  float linv = 1.0f / lrow;
  float lr[8];
#pragma unroll
  for (int r = 0; r < 8; ++r) lr[r] = __shfl(linv, r + 8 * half, 32);
#pragma unroll
  for (int blk = 0; blk < 8; ++blk)
#pragma unroll
    for (int r = 0; r < 8; ++r) {
      int rowm = mt * 16 + r + 8 * half;
      int col  = h * HEAD_D + blk * 16 + l15;
      O[(size_t)(b * S_LEN + rowm) * D_MODELC + col] = f2bf(oacc[blk][r] * lr[r]);
    }
}

// ----------------------------------------------------------------- launch ---
extern "C" void kernel_launch(void* const* d_in, const int* in_sizes, int n_in,
                              void* d_out, int out_size, void* d_ws, size_t ws_size,
                              hipStream_t stream) {
  const float* x  = (const float*)d_in[0];
  const float* Wq = (const float*)d_in[1];
  const float* Wk = (const float*)d_in[2];
  const float* Wv = (const float*)d_in[3];
  const float* Wo = (const float*)d_in[4];

  const int BS = BATCH * S_LEN;   // 4096 total rows

  unsigned short* ws = (unsigned short*)d_ws;
  size_t off = 0;
  unsigned short* xb  = ws + off; off += (size_t)BS * D_MODELC;
  unsigned short* wqb = ws + off; off += (size_t)D_MODELC * D_MODELC;
  unsigned short* wkb = ws + off; off += (size_t)HEAD_D * D_MODELC;
  unsigned short* wvb = ws + off; off += (size_t)HEAD_D * D_MODELC;
  unsigned short* wob = ws + off; off += (size_t)D_MODELC * D_MODELC;
  unsigned short* qb  = ws + off; off += (size_t)BS * D_MODELC;
  unsigned short* kb  = ws + off; off += (size_t)BS * HEAD_D;
  unsigned short* vtb = ws + off; off += (size_t)BS * HEAD_D;   // [d][B*S]
  unsigned short* ao  = ws + off; off += (size_t)BS * D_MODELC;

  f32_to_bf16_kernel<<<(BS * D_MODELC) / 256, 256, 0, stream>>>(x, xb, BS * D_MODELC);
  f32_to_bf16_kernel<<<(D_MODELC * D_MODELC) / 256, 256, 0, stream>>>(Wq, wqb, D_MODELC * D_MODELC);
  f32_to_bf16_kernel<<<(HEAD_D * D_MODELC) / 256, 256, 0, stream>>>(Wk, wkb, HEAD_D * D_MODELC);
  f32_to_bf16_kernel<<<(HEAD_D * D_MODELC) / 256, 256, 0, stream>>>(Wv, wvb, HEAD_D * D_MODELC);
  f32_to_bf16_kernel<<<(D_MODELC * D_MODELC) / 256, 256, 0, stream>>>(Wo, wob, D_MODELC * D_MODELC);

  const float qscale = 0.088388347648318447f;  // 1/sqrt(head_dim)

  // Q = (x @ Wq^T) * qscale   [bf16]
  {
    int waves = (BS / 32) * (D_MODELC / 64);
    gemm_bf16_wmma<<<waves / 4, 128, 0, stream>>>(xb, wqb, qb, BS, D_MODELC,
                                                  D_MODELC, qscale, 1);
  }
  // K = x @ Wk^T   [bf16]
  {
    int waves = (BS / 32) * (HEAD_D / 64);
    gemm_bf16_wmma<<<waves / 4, 128, 0, stream>>>(xb, wkb, kb, BS, HEAD_D,
                                                  D_MODELC, 1.0f, 1);
  }
  // V^T = (x @ Wv^T)^T   [bf16 transposed store -> [d][B*S]]
  {
    int waves = (BS / 32) * (HEAD_D / 64);
    gemm_bf16_wmma<<<waves / 4, 128, 0, stream>>>(xb, wvb, vtb, BS, HEAD_D,
                                                  D_MODELC, 1.0f, 2);
  }
  // flash MQA: 2 * 16 * 128 = 4096 waves, 4 waves/block sharing K/V via TDM
  {
    int waves = BATCH * NUM_H * (S_LEN / 16);
    mqa_flash_attn<<<waves / 4, 128, 0, stream>>>(qb, kb, vtb, ao);
  }
  // out = attnOut @ Wo^T   [f32 to d_out]
  {
    int waves = (BS / 32) * (D_MODELC / 64);
    gemm_bf16_wmma<<<waves / 4, 128, 0, stream>>>(ao, wob, (float*)d_out, BS,
                                                  D_MODELC, D_MODELC, 1.0f, 0);
  }
}
